// VectorQuantizer_58411555225656
// MI455X (gfx1250) — matrix operand order repaired and checked by reading, not compiled, not run on hardware
//
#include <hip/hip_runtime.h>
#include <stdint.h>

typedef __attribute__((ext_vector_type(16))) __bf16   v16bf;
typedef __attribute__((ext_vector_type(8)))  float    v8f;
typedef __attribute__((ext_vector_type(8)))  uint32_t v8u;

namespace {
constexpr int kD         = 64;
constexpr int kK         = 512;
constexpr int kHW        = 64 * 64;        // 4096 pixels per image plane
constexpr int kImgStride = kD * kHW;       // 262144 floats per image
constexpr int kNPix      = 32 * kHW;       // 131072 pixel rows
constexpr int kNTiles    = kNPix / 16;     // 8192 tiles of 16 pixels
constexpr int kWavesPerBlock = 8;
constexpr int kTilesPerWave  = 4;          // processed as 2 pairs of 2 tiles
constexpr int kMainBlocks = kNTiles / (kWavesPerBlock * kTilesPerWave); // 256
constexpr int kBFragDwords = 32 /*ntile*/ * 2 /*kchunk*/ * 32 /*lane*/ * 8; // 16384 dwords = 64KB
constexpr uint32_t kIdxMask = 511u;        // 9-bit code index packed in mantissa LSBs
}

// ---------------------------------------------------------------------------
// Stage 1a: repack codebook [D=64, K=512] f32 (row-major, stride K) into bf16
// WMMA B-fragments. Slot s = (t, chunk, lane); v16bf elem i of lane l holds
// B[K = (l>>4)*16 + i][N = l&15] with global d = chunk*32 + K, col = t*16 + N.
// ---------------------------------------------------------------------------
__global__ void vq_prep_b(const float* __restrict__ cb, uint32_t* __restrict__ wsB) {
  const int s    = blockIdx.x * blockDim.x + threadIdx.x;   // 0..2047
  const int t    = s >> 6;
  const int chnk = (s >> 5) & 1;
  const int lane = s & 31;
  const int hf   = lane >> 4;
  const int col  = t * 16 + (lane & 15);
  uint32_t w[8];
#pragma unroll
  for (int j = 0; j < 8; ++j) {
    const int d0 = chnk * 32 + hf * 16 + 2 * j;
    const __bf16 b0 = (__bf16)cb[d0 * kK + col];
    const __bf16 b1 = (__bf16)cb[(d0 + 1) * kK + col];
    w[j] = (uint32_t)__builtin_bit_cast(unsigned short, b0) |
           ((uint32_t)__builtin_bit_cast(unsigned short, b1) << 16);
  }
  uint4* dst = (uint4*)(wsB + (size_t)s * 8);
  dst[0] = make_uint4(w[0], w[1], w[2], w[3]);
  dst[1] = make_uint4(w[4], w[5], w[6], w[7]);
}

// Stage 1b: c2[k] = sum_d codebook[d][k]^2 + 1.0f. The +1 bias guarantees
// score = 1 + ||c||^2 - 2 x.c > 0 (|2 x.c| <= 2||x|| ||c|| << 1), so packed
// scores order identically as unsigned integers -> argmin via v_min_u32.
__global__ void vq_prep_c2(const float* __restrict__ cb, float* __restrict__ wsC2) {
  const int k = blockIdx.x * blockDim.x + threadIdx.x;
  if (k >= kK) return;
  float acc = 1.0f;
#pragma unroll 8
  for (int d = 0; d < kD; ++d) {
    const float v = cb[d * kK + k];
    acc = fmaf(v, v, acc);
  }
  wsC2[k] = acc;
}

// ---------------------------------------------------------------------------
// Main: per wave iteration, a PAIR of 16-pixel tiles (32 pixels) so every
// B-fragment ds_load feeds two WMMAs. score[m][k] = 1 + ||c_k||^2 - 2 x_m.c_k
// via D = A(-2x bf16) x B(cb bf16) + C(1 + c2). Argmin: pack 9-bit code index
// into the mantissa LSBs of the (positive) f32 score and reduce with unsigned
// integer min (bit order == float order for positives; lower index wins ties;
// perturbation 2^-14 << score gaps).
// ---------------------------------------------------------------------------
__global__ void __launch_bounds__(256)
vq_main(const float* __restrict__ enc, const float* __restrict__ cb,
        const uint32_t* __restrict__ wsB, const float* __restrict__ wsC2,
        float* __restrict__ out) {
  __shared__ __align__(32) uint32_t ldsB[kBFragDwords];  // 64KB bf16 B-fragments
  __shared__ float ldsC2[kK];                            // 2KB
  __shared__ int   ldsIdx[kWavesPerBlock][32];

  const int tid = threadIdx.x;

  // Stage block-shared B fragments + c2 into LDS (coalesced b128 copies).
  {
    const uint4* src = (const uint4*)wsB;
    uint4*       dst = (uint4*)ldsB;
#pragma unroll
    for (int i = 0; i < kBFragDwords / 4 / 256; ++i)  // 16 iters
      dst[tid + i * 256] = src[tid + i * 256];
    ldsC2[tid]       = wsC2[tid];
    ldsC2[tid + 256] = wsC2[tid + 256];
  }
  __syncthreads();

  const int wid  = tid >> 5;
  const int lane = tid & 31;
  const int m    = lane & 15;   // pixel row within tile / N column within 16
  const int hf   = lane >> 4;   // half-group select
  const v16bf* bfrags = (const v16bf*)ldsB;

  for (int ip = 0; ip < kTilesPerWave / 2; ++ip) {
    const int T0    = (blockIdx.x * kWavesPerBlock + wid) * kTilesPerWave + ip * 2;
    const int pbase = T0 * 16;        // pixels pbase..pbase+31: same image, same h row
    const int bimg  = pbase >> 12;
    const int hw    = pbase & (kHW - 1);
    const float* encp = enc + (size_t)bimg * kImgStride + hw + m;

    // A fragments for both subtiles: elem i -> row m, K = hf*16 + i.
    v16bf a0, a1, a2, a3;
#pragma unroll
    for (int i = 0; i < 16; ++i) {
      const int d = hf * 16 + i;
      a0[i] = (__bf16)(-2.0f * encp[(size_t)d * kHW]);
      a1[i] = (__bf16)(-2.0f * encp[(size_t)(d + 32) * kHW]);
      a2[i] = (__bf16)(-2.0f * encp[(size_t)d * kHW + 16]);
      a3[i] = (__bf16)(-2.0f * encp[(size_t)(d + 32) * kHW + 16]);
    }

    v8u best0, best1;
#pragma unroll
    for (int r = 0; r < 8; ++r) { best0[r] = 0x7F000000u; best1[r] = 0x7F000000u; }

    for (int t = 0; t < 32; ++t) {
      const v16bf b0 = bfrags[(t * 2 + 0) * 32 + lane];
      const v16bf b1 = bfrags[(t * 2 + 1) * 32 + lane];
      const float c2v = ldsC2[t * 16 + m];
      v8f c;
#pragma unroll
      for (int r = 0; r < 8; ++r) c[r] = c2v;
      v8f s0 = __builtin_amdgcn_wmma_f32_16x16x32_bf16(false, a0, false, b0,
                                                       (short)0, c, false, false);
      s0 = __builtin_amdgcn_wmma_f32_16x16x32_bf16(false, a1, false, b1,
                                                   (short)0, s0, false, false);
      v8f s1 = __builtin_amdgcn_wmma_f32_16x16x32_bf16(false, a2, false, b0,
                                                       (short)0, c, false, false);
      s1 = __builtin_amdgcn_wmma_f32_16x16x32_bf16(false, a3, false, b1,
                                                   (short)0, s1, false, false);
      // Whole-vector pack + unsigned min: 8x v_and_or_b32 + 8x v_min_u32 each.
      const uint32_t kcol = (uint32_t)(t * 16 + m);
      const v8u q0 = (__builtin_bit_cast(v8u, s0) & ~kIdxMask) | kcol;
      const v8u q1 = (__builtin_bit_cast(v8u, s1) & ~kIdxMask) | kcol;
      best0 = __builtin_elementwise_min(best0, q0);
      best1 = __builtin_elementwise_min(best1, q1);
    }

    // Cross-lane unsigned-min within each 16-lane half-group (rows r / 8+r).
#pragma unroll
    for (int msk = 8; msk >= 1; msk >>= 1) {
#pragma unroll
      for (int r = 0; r < 8; ++r) {
        const uint32_t a0v = best0[r];
        const uint32_t a1v = best1[r];
        const uint32_t o0 = (uint32_t)__shfl_xor((int)a0v, msk, 32);
        const uint32_t o1 = (uint32_t)__shfl_xor((int)a1v, msk, 32);
        best0[r] = (o0 < a0v) ? o0 : a0v;
        best1[r] = (o1 < a1v) ? o1 : a1v;
      }
    }

    if (m == 0) {  // lane 0 -> rows 0..7, lane 16 -> rows 8..15, per subtile
#pragma unroll
      for (int r = 0; r < 8; ++r) {
        ldsIdx[wid][hf * 8 + r]      = (int)(best0[r] & kIdxMask);
        ldsIdx[wid][16 + hf * 8 + r] = (int)(best1[r] & kIdxMask);
      }
    }
    __syncthreads();

    // Scatter quantized vectors back into [B, D, H, W]; lane handles pixel m
    // of each subtile, half-groups split the d range (contiguous 64B stores).
#pragma unroll
    for (int s = 0; s < 2; ++s) {
      const int myk = ldsIdx[wid][s * 16 + m];
      float* outp = out + (size_t)bimg * kImgStride + hw + s * 16 + m;
#pragma unroll 8
      for (int j = 0; j < 32; ++j) {
        const int d = hf * 32 + j;
        outp[(size_t)d * kHW] = cb[d * kK + myk];
      }
    }
  }
}

extern "C" void kernel_launch(void* const* d_in, const int* in_sizes, int n_in,
                              void* d_out, int out_size, void* d_ws, size_t ws_size,
                              hipStream_t stream) {
  const float* enc = (const float*)d_in[0];   // [32, 64, 64, 64] f32
  const float* cb  = (const float*)d_in[1];   // [64, 512] f32
  float* out = (float*)d_out;                 // [32, 64, 64, 64] f32
  uint32_t* wsB = (uint32_t*)d_ws;                      // 64KB bf16 B fragments
  float*    wsC2 = (float*)((char*)d_ws + 65536);       // 2KB 1 + ||c||^2

  vq_prep_b<<<8, 256, 0, stream>>>(cb, wsB);
  vq_prep_c2<<<2, 256, 0, stream>>>(cb, wsC2);
  vq_main<<<kMainBlocks, 256, 0, stream>>>(enc, cb, wsB, wsC2, out);
}
